// AttentionBlock_2637109920520
// MI455X (gfx1250) — matrix operand order repaired and checked by reading, not compiled, not run on hardware
//
#include <hip/hip_runtime.h>

#define NUM_CH 256
#define NHEAD  4
#define DK     64
#define BATCH  16
#define SEQ    1024            // H*W = 32*32
#define PROJ   (NHEAD*DK*3)    // 768

typedef __attribute__((ext_vector_type(16))) _Float16 v16h;
typedef __attribute__((ext_vector_type(8)))  _Float16 v8h;
typedef __attribute__((ext_vector_type(8)))  float    v8f;

union V16 { v16h v; v8h h[2]; };

// ---------------------------------------------------------------------------
// WMMA helper: D = A(16x32 f16) * B(32x16 f16) + C(16x16 f32)
// ---------------------------------------------------------------------------
__device__ inline v8f wmma_f16(v16h a, v16h b, v8f c) {
  return __builtin_amdgcn_wmma_f32_16x16x32_f16(
      /*neg_a=*/false, a, /*neg_b=*/false, b,
      /*c_mod=*/(short)0, c, /*reuse_a=*/false, /*reuse_b=*/false);
}

// A-operand (16x32, f16), ISA 7.12.2 layout:
//   lane L holds row M = L&15.
//   halves 0..7  -> K = k0 + (L<16 ? 0 : 8)  + i      (contiguous 16B)
//   halves 8..15 -> K = k0 + 16 + (L<16?0:8) + (i-8)  (contiguous 16B)
__device__ inline v16h load_a(const _Float16* __restrict__ Amat, int ld, int k0) {
  int lane = threadIdx.x & 31;
  int g = lane >> 4;
  const _Float16* base = Amat + (size_t)(lane & 15) * ld + k0 + g * 8;
  V16 u;
  u.h[0] = *(const v8h*)(base);
  u.h[1] = *(const v8h*)(base + 16);
  return u.v;
}

// B-operand (32x16, f16), column-major striping:
//   lane L holds column N = n0 + (L&15).
//   halves 0..7  -> K = k0 + (L<16 ? 0 : 8)  + i      (contiguous 16B)
//   halves 8..15 -> K = k0 + 16 + (L<16?0:8) + (i-8)  (contiguous 16B)
__device__ inline v16h load_b(const _Float16* __restrict__ Bm, int ld, int n0, int k0) {
  int lane = threadIdx.x & 31;
  int g = lane >> 4;
  const _Float16* base = Bm + (size_t)(n0 + (lane & 15)) * ld + k0 + g * 8;
  V16 u;
  u.h[0] = *(const v8h*)(base);
  u.h[1] = *(const v8h*)(base + 16);
  return u.v;
}

// ---------------------------------------------------------------------------
// Prep: xt16[b][n][c] = (f16) x[b][c][n]   (transpose [C,N] -> [N,C] + cvt)
// ---------------------------------------------------------------------------
__global__ void prep_x_kernel(const float* __restrict__ x, _Float16* __restrict__ xt16) {
  size_t idx = (size_t)blockIdx.x * blockDim.x + threadIdx.x; // linear over [b][c][n]
  int n = (int)(idx % SEQ);
  int c = (int)((idx / SEQ) % NUM_CH);
  int b = (int)(idx / ((size_t)SEQ * NUM_CH));
  xt16[((size_t)b * SEQ + n) * NUM_CH + c] = (_Float16)x[idx];
}

__global__ void cvt_f16_kernel(const float* __restrict__ src, _Float16* __restrict__ dst, int n) {
  int i = blockIdx.x * blockDim.x + threadIdx.x;
  if (i < n) dst[i] = (_Float16)src[i];
}

// ---------------------------------------------------------------------------
// QKV projection: qkv[b,n,:] = xt[b,n,:] @ Wp^T + bp
// One wave computes a 32(n) x 64(o) tile: every B-operand feeds 2 WMMAs.
// Q,K stored [b,h,n,d]; V stored transposed [b,h,d,n] for the PV B-operand.
// ---------------------------------------------------------------------------
__global__ __launch_bounds__(32)
void qkv_gemm_kernel(const _Float16* __restrict__ xt16, const _Float16* __restrict__ Wp16,
                     const float* __restrict__ bp,
                     _Float16* __restrict__ q16, _Float16* __restrict__ k16,
                     _Float16* __restrict__ vt16) {
  int id = blockIdx.x;
  int ot = id % (PROJ / 64);                    // 12 o-tiles
  int nt = (id / (PROJ / 64)) % (SEQ / 32);     // 32 n-tiles
  int b  = id / ((PROJ / 64) * (SEQ / 32));
  int o0 = ot * 64, n0 = nt * 32;

  const _Float16* A0 = xt16 + ((size_t)b * SEQ + n0) * NUM_CH;
  const _Float16* A1 = A0 + (size_t)16 * NUM_CH;
  v8f acc[2][4] = {};
  for (int k0 = 0; k0 < NUM_CH; k0 += 32) {
    __builtin_prefetch(A0 + k0 + 32, 0, 1);     // global_prefetch_b8 next k-chunk
    v16h a0 = load_a(A0, NUM_CH, k0);
    v16h a1 = load_a(A1, NUM_CH, k0);
#pragma unroll
    for (int c = 0; c < 4; ++c) {
      v16h bop = load_b(Wp16, NUM_CH, o0 + c * 16, k0);
      acc[0][c] = wmma_f16(a0, bop, acc[0][c]);
      acc[1][c] = wmma_f16(a1, bop, acc[1][c]);
    }
  }
  int lane = threadIdx.x & 31, g = lane >> 4, cn = lane & 15;
#pragma unroll
  for (int c = 0; c < 4; ++c) {
    int o = o0 + c * 16 + cn;
    int h = o / (3 * DK);
    int t = o % (3 * DK);
    float bias = bp[o];
    size_t bh = (size_t)b * NHEAD + h;
#pragma unroll
    for (int rt = 0; rt < 2; ++rt) {
#pragma unroll
      for (int r = 0; r < 8; ++r) {
        int n = n0 + rt * 16 + r + 8 * g;
        float val = acc[rt][c][r] + bias;
        if (t < DK)            q16[(bh * SEQ + n) * DK + t]             = (_Float16)val;
        else if (t < 2 * DK)   k16[(bh * SEQ + n) * DK + (t - DK)]      = (_Float16)val;
        else                   vt16[(bh * DK + (t - 2 * DK)) * SEQ + n] = (_Float16)val;
      }
    }
  }
}

// ---------------------------------------------------------------------------
// Flash attention: one wave owns 32 query rows of one (b,h); streams all 1024
// keys in chunks of 32 with online softmax. K/V B-operands are shared across
// both 16-row sub-tiles (2x operand reuse). S and O accumulated in f32.
// ---------------------------------------------------------------------------
__global__ __launch_bounds__(32)
void attn_kernel(const _Float16* __restrict__ q16, const _Float16* __restrict__ k16,
                 const _Float16* __restrict__ vt16, _Float16* __restrict__ ao16) {
  __shared__ _Float16 ptile[32 * 32];           // P tile: C-layout -> A-layout bridge

  int id = blockIdx.x;
  int nt = id % (SEQ / 32);
  int h  = (id / (SEQ / 32)) % NHEAD;
  int b  = id / ((SEQ / 32) * NHEAD);
  int n0 = nt * 32;
  size_t bh = (size_t)b * NHEAD + h;

  const _Float16* Q  = q16  + (bh * SEQ + n0) * DK;
  const _Float16* K  = k16  + bh * SEQ * DK;
  const _Float16* VT = vt16 + bh * DK * SEQ;

  int lane = threadIdx.x & 31, g = lane >> 4, cn = lane & 15;

  v16h aq[2][2];                                // [row-subtile][d-chunk]
#pragma unroll
  for (int rt = 0; rt < 2; ++rt) {
    aq[rt][0] = load_a(Q + (size_t)rt * 16 * DK, DK, 0);
    aq[rt][1] = load_a(Q + (size_t)rt * 16 * DK, DK, 32);
  }

  v8f O[2][4] = {};
  float mrow[2][8], lrow[2][8];
#pragma unroll
  for (int rt = 0; rt < 2; ++rt)
#pragma unroll
    for (int r = 0; r < 8; ++r) { mrow[rt][r] = -1e30f; lrow[rt][r] = 0.0f; }
  const float scale = 0.125f;                   // DK^-0.5

  for (int j0 = 0; j0 < SEQ; j0 += 32) {
    // K B-operands for keys [j0, j0+32): loaded once, used by both row tiles
    v16h kb00 = load_b(K, DK, j0,      0);
    v16h kb01 = load_b(K, DK, j0,      32);
    v16h kb10 = load_b(K, DK, j0 + 16, 0);
    v16h kb11 = load_b(K, DK, j0 + 16, 32);

#pragma unroll
    for (int rt = 0; rt < 2; ++rt) {
      v8f s0 = {}, s1 = {};
      s0 = wmma_f16(aq[rt][0], kb00, s0);
      s0 = wmma_f16(aq[rt][1], kb01, s0);
      s1 = wmma_f16(aq[rt][0], kb10, s1);
      s1 = wmma_f16(aq[rt][1], kb11, s1);

      // Online softmax. Row m = r + 8g lives in VGPR r across a 16-lane half,
      // so reductions are shfl-xor over masks {1,2,4,8} (stay inside the half).
#pragma unroll
      for (int r = 0; r < 8; ++r) {
        float x0 = s0[r] * scale, x1 = s1[r] * scale;
        float vmax = fmaxf(x0, x1);
        vmax = fmaxf(vmax, __shfl_xor(vmax, 1, 32));
        vmax = fmaxf(vmax, __shfl_xor(vmax, 2, 32));
        vmax = fmaxf(vmax, __shfl_xor(vmax, 4, 32));
        vmax = fmaxf(vmax, __shfl_xor(vmax, 8, 32));
        float mn    = fmaxf(mrow[rt][r], vmax);
        float alpha = __expf(mrow[rt][r] - mn);
        float p0 = __expf(x0 - mn);
        float p1 = __expf(x1 - mn);
        float rs = p0 + p1;
        rs += __shfl_xor(rs, 1, 32);
        rs += __shfl_xor(rs, 2, 32);
        rs += __shfl_xor(rs, 4, 32);
        rs += __shfl_xor(rs, 8, 32);
        lrow[rt][r] = lrow[rt][r] * alpha + rs;
        mrow[rt][r] = mn;
#pragma unroll
        for (int c = 0; c < 4; ++c) O[rt][c][r] *= alpha;
        int m = rt * 16 + r + 8 * g;
        ptile[m * 32 + cn]      = (_Float16)p0;
        ptile[m * 32 + 16 + cn] = (_Float16)p1;
      }
    }
    __syncthreads();

    // O(32x64) += P(32x32) @ V(32x64); each V B-operand feeds 2 WMMAs
    v16h ap0 = load_a(ptile,           32, 0);
    v16h ap1 = load_a(ptile + 16 * 32, 32, 0);
#pragma unroll
    for (int c = 0; c < 4; ++c) {
      v16h bv = load_b(VT, SEQ, c * 16, j0);
      O[0][c] = wmma_f16(ap0, bv, O[0][c]);
      O[1][c] = wmma_f16(ap1, bv, O[1][c]);
    }
    __syncthreads();
  }

  // Normalize and emit attention output as f16 [b, n, h*64 + d]
#pragma unroll
  for (int rt = 0; rt < 2; ++rt) {
#pragma unroll
    for (int c = 0; c < 4; ++c) {
#pragma unroll
      for (int r = 0; r < 8; ++r) {
        int n = n0 + rt * 16 + r + 8 * g;
        float val = O[rt][c][r] / lrow[rt][r];
        ao16[((size_t)b * SEQ + n) * NUM_CH + h * DK + c * 16 + cn] = (_Float16)val;
      }
    }
  }
}

// ---------------------------------------------------------------------------
// Output projection + bias + residual, transposed back to [B, C, H, W] (f32)
// 32(n) x 64(c) tile per wave, B-operands reused across both row sub-tiles.
// ---------------------------------------------------------------------------
__global__ __launch_bounds__(32)
void out_gemm_kernel(const _Float16* __restrict__ ao16, const _Float16* __restrict__ Wo16,
                     const float* __restrict__ bo, const float* __restrict__ x,
                     float* __restrict__ out) {
  int id = blockIdx.x;
  int ct = id % (NUM_CH / 64);                  // 4 c-tiles
  int nt = (id / (NUM_CH / 64)) % (SEQ / 32);
  int b  = id / ((NUM_CH / 64) * (SEQ / 32));
  int c0 = ct * 64, n0 = nt * 32;

  const _Float16* A0 = ao16 + ((size_t)b * SEQ + n0) * NUM_CH;
  const _Float16* A1 = A0 + (size_t)16 * NUM_CH;
  v8f acc[2][4] = {};
  for (int k0 = 0; k0 < NUM_CH; k0 += 32) {
    __builtin_prefetch(A0 + k0 + 32, 0, 1);
    v16h a0 = load_a(A0, NUM_CH, k0);
    v16h a1 = load_a(A1, NUM_CH, k0);
#pragma unroll
    for (int c = 0; c < 4; ++c) {
      v16h bop = load_b(Wo16, NUM_CH, c0 + c * 16, k0);
      acc[0][c] = wmma_f16(a0, bop, acc[0][c]);
      acc[1][c] = wmma_f16(a1, bop, acc[1][c]);
    }
  }
  int lane = threadIdx.x & 31, g = lane >> 4, cn = lane & 15;
#pragma unroll
  for (int c = 0; c < 4; ++c) {
    int ch = c0 + c * 16 + cn;
    float bias = bo[ch];
#pragma unroll
    for (int rt = 0; rt < 2; ++rt) {
#pragma unroll
      for (int r = 0; r < 8; ++r) {
        int n = n0 + rt * 16 + r + 8 * g;
        size_t idx = ((size_t)b * NUM_CH + ch) * SEQ + n;  // [b][c][n] == [B,C,H,W]
        out[idx] = acc[rt][c][r] + bias + x[idx];          // residual from original x
      }
    }
  }
}

// ---------------------------------------------------------------------------
extern "C" void kernel_launch(void* const* d_in, const int* in_sizes, int n_in,
                              void* d_out, int out_size, void* d_ws, size_t ws_size,
                              hipStream_t stream) {
  const float* x  = (const float*)d_in[0];   // [16,256,32,32]
  const float* Wp = (const float*)d_in[1];   // [768,256]
  const float* bp = (const float*)d_in[2];   // [768]
  const float* Wo = (const float*)d_in[3];   // [256,256]
  const float* bo = (const float*)d_in[4];   // [256]
  float* out = (float*)d_out;

  char* ws = (char*)d_ws;
  const size_t SZ_XT  = (size_t)BATCH * SEQ * NUM_CH * 2;      // 8 MB
  const size_t SZ_WP  = (size_t)PROJ * NUM_CH * 2;             // 384 KB
  const size_t SZ_WO  = (size_t)NUM_CH * NUM_CH * 2;           // 128 KB
  const size_t SZ_QKV = (size_t)BATCH * NHEAD * SEQ * DK * 2;  // 8 MB each
  _Float16* xt16 = (_Float16*)(ws);                 ws += SZ_XT;
  _Float16* Wp16 = (_Float16*)(ws);                 ws += SZ_WP;
  _Float16* Wo16 = (_Float16*)(ws);                 ws += SZ_WO;
  _Float16* q16  = (_Float16*)(ws);                 ws += SZ_QKV;
  _Float16* k16  = (_Float16*)(ws);                 ws += SZ_QKV;
  _Float16* vt16 = (_Float16*)(ws);                 ws += SZ_QKV;
  _Float16* ao16 = (_Float16*)(ws);                 ws += SZ_XT;

  // 1. transpose + convert x; convert weights to f16
  prep_x_kernel<<<(BATCH * NUM_CH * SEQ) / 256, 256, 0, stream>>>(x, xt16);
  cvt_f16_kernel<<<(PROJ * NUM_CH + 255) / 256, 256, 0, stream>>>(Wp, Wp16, PROJ * NUM_CH);
  cvt_f16_kernel<<<(NUM_CH * NUM_CH + 255) / 256, 256, 0, stream>>>(Wo, Wo16, NUM_CH * NUM_CH);

  // 2. QKV projection (WMMA), V stored transposed
  qkv_gemm_kernel<<<BATCH * (SEQ / 32) * (PROJ / 64), 32, 0, stream>>>(
      xt16, Wp16, bp, q16, k16, vt16);

  // 3. flash attention (WMMA + online softmax)
  attn_kernel<<<BATCH * NHEAD * (SEQ / 32), 32, 0, stream>>>(q16, k16, vt16, ao16);

  // 4. output projection + bias + residual + transpose back (WMMA)
  out_gemm_kernel<<<BATCH * (SEQ / 32) * (NUM_CH / 64), 32, 0, stream>>>(
      ao16, Wo16, bo, x, out);
}
